// DecoderWithBahdanauAttention_5712306504416
// MI455X (gfx1250) — compile-verified, matrix operand order checked by reading, not run on hardware
//
#include <hip/hip_runtime.h>
#include <hip/hip_bf16.h>

// ---------------------------------------------------------------------------
// Bahdanau-attention GRU decoder for MI455X (gfx1250).
// Strategy: pre-convert all weights to bf16 in workspace (~90MB, fits 192MB L2),
// run every GEMM with v_wmma_f32_16x16x32_bf16 (fp32 accumulate), software-
// pipelined fragment loads, and a 32x64 per-wave output tile (MT=2 x NT=4) so
// each streamed B fragment feeds two WMMAs and weights are read once per step.
// Tiny attention-softmax / GRU-gate / log-softmax parts stay in VALU kernels.
// Workspace use: ~116 MB.
// ---------------------------------------------------------------------------

typedef __attribute__((ext_vector_type(16))) __bf16 v16bf;
typedef __attribute__((ext_vector_type(8)))  float  v8f;

static constexpr int V = 32000, H = 1024, A = 1024;
static constexpr int B = 32, S = 64, T = 64;

union FragBF { v16bf v; uint4 q[2]; };

__device__ __forceinline__ unsigned short f2bf(float f) {
  unsigned int u = __float_as_uint(f);
  unsigned int r = (u + 0x7FFFu + ((u >> 16) & 1u)) >> 16;   // round-to-nearest-even
  return (unsigned short)r;
}

// ---------------------------------------------------------------------------
// Generic bf16 WMMA GEMM:  C[M,N] = act( Aact[M,K] * W[N,K]^T + bias[N] )
// Aact row-major bf16 (stride lda), W row-major [N,K] bf16 (stride ldw).
// One wave per block; each wave owns a (16*MT) x (16*NT) = 32x64 output tile.
// K-loop is software-pipelined: fragments for step k+32 are loaded while the
// WMMAs for step k consume already-resident registers. Each B fragment is
// reused across MT row-tiles, each A fragment across NT col-tiles.
// grid = (N/(16*NT), M/(16*MT)), block = 32.
// ---------------------------------------------------------------------------
#define NT 4
#define MT 2
__global__ void __launch_bounds__(32)
gemm_bf16_wmma(const unsigned short* __restrict__ Aact, int lda,
               const unsigned short* __restrict__ W,   int ldw,
               const float* __restrict__ bias,
               float* __restrict__ C, int ldc,
               unsigned short* __restrict__ Cbf,
               int K, int act_tanh)
{
  const int lane   = threadIdx.x;
  const int laneLo = lane & 15;
  const int hi     = lane >> 4;                       // 0: K {0..7,16..23}, 1: K {8..15,24..31}
  const long tileN = (long)blockIdx.x * (16 * NT);
  const long tileM = (long)blockIdx.y * (16 * MT);

  v8f c[MT][NT] = {};                                 // fp32 accumulators

  // hoisted per-lane base pointers (indexed by +k inside the loop)
  const unsigned short* Ap[MT];
#pragma unroll
  for (int m = 0; m < MT; ++m)
    Ap[m] = Aact + (tileM + m * 16 + laneLo) * (long)lda + hi * 8;
  const unsigned short* Wp[NT];
#pragma unroll
  for (int i = 0; i < NT; ++i)
    Wp[i] = W + (tileN + i * 16 + laneLo) * (long)ldw + hi * 8;

  // prologue: load fragments for k = 0
  FragBF a_cur[MT], b_cur[NT];
#pragma unroll
  for (int m = 0; m < MT; ++m) {
    a_cur[m].q[0] = *(const uint4*)(Ap[m]);
    a_cur[m].q[1] = *(const uint4*)(Ap[m] + 16);
  }
#pragma unroll
  for (int i = 0; i < NT; ++i) {
    b_cur[i].q[0] = *(const uint4*)(Wp[i]);
    b_cur[i].q[1] = *(const uint4*)(Wp[i] + 16);
  }

#pragma unroll 2
  for (int k = 32; k < K; k += 32) {
    // issue next iteration's loads first (independent registers)
    FragBF a_nxt[MT], b_nxt[NT];
#pragma unroll
    for (int m = 0; m < MT; ++m) {
      a_nxt[m].q[0] = *(const uint4*)(Ap[m] + k);
      a_nxt[m].q[1] = *(const uint4*)(Ap[m] + k + 16);
    }
#pragma unroll
    for (int i = 0; i < NT; ++i) {
      b_nxt[i].q[0] = *(const uint4*)(Wp[i] + k);
      b_nxt[i].q[1] = *(const uint4*)(Wp[i] + k + 16);
      __builtin_prefetch(Wp[i] + k + 128, 0, 1);      // stream W ahead (256B)
    }
    // compute on the already-resident fragments
#pragma unroll
    for (int i = 0; i < NT; ++i)
#pragma unroll
      for (int m = 0; m < MT; ++m)
        c[m][i] = __builtin_amdgcn_wmma_f32_16x16x32_bf16(
                      false, a_cur[m].v, false, b_cur[i].v,
                      (short)0, c[m][i], false, false);
#pragma unroll
    for (int m = 0; m < MT; ++m) a_cur[m] = a_nxt[m];
#pragma unroll
    for (int i = 0; i < NT; ++i) b_cur[i] = b_nxt[i];
  }
  // epilogue: last K block
#pragma unroll
  for (int i = 0; i < NT; ++i)
#pragma unroll
    for (int m = 0; m < MT; ++m)
      c[m][i] = __builtin_amdgcn_wmma_f32_16x16x32_bf16(
                    false, a_cur[m].v, false, b_cur[i].v,
                    (short)0, c[m][i], false, false);

  // D layout: lanes 0-15 -> N=lane, M=r; lanes 16-31 -> N=lane-16, M=r+8
#pragma unroll
  for (int m = 0; m < MT; ++m) {
    const long mbase = tileM + m * 16 + hi * 8;
#pragma unroll
    for (int i = 0; i < NT; ++i) {
      const long n = tileN + i * 16 + laneLo;
      const float bv = bias ? bias[n] : 0.0f;
#pragma unroll
      for (int r = 0; r < 8; ++r) {
        float v = c[m][i][r] + bv;
        if (act_tanh) v = tanhf(v);
        const long idx = (mbase + r) * (long)ldc + n;
        C[idx] = v;
        if (Cbf) Cbf[idx] = f2bf(v);
      }
    }
  }
}

// ---------------------------------------------------------------------------
// One-time prep kernels
// ---------------------------------------------------------------------------
__global__ void cvt_bf16(const float* __restrict__ src, unsigned short* __restrict__ dst, long n) {
  long i = (long)blockIdx.x * blockDim.x + threadIdx.x;
  if (i < n) dst[i] = f2bf(src[i]);
}

// split w_att[A,2H] into We[A,H] (cols 0..H-1) and Wd[A,H] (cols H..2H-1), bf16
__global__ void split_watt(const float* __restrict__ w_att,
                           unsigned short* __restrict__ We, unsigned short* __restrict__ Wd) {
  long i = (long)blockIdx.x * blockDim.x + threadIdx.x;
  if (i >= (long)A * H) return;
  long row = i >> 10, col = i & (H - 1);
  We[i] = f2bf(w_att[row * (2 * H) + col]);
  Wd[i] = f2bf(w_att[row * (2 * H) + H + col]);
}

// emb_x[b,t,:] = bf16(embedding[target[b,t]])
__global__ void gather_emb(const float* __restrict__ emb, const int* __restrict__ tok,
                           unsigned short* __restrict__ dst) {
  long i = (long)blockIdx.x * blockDim.x + threadIdx.x;
  if (i >= (long)B * T * H) return;
  long bt = i >> 10, col = i & (H - 1);
  dst[i] = f2bf(emb[(long)tok[bt] * H + col]);
}

__global__ void init_h(const float* __restrict__ h0, float* __restrict__ h,
                       unsigned short* __restrict__ h_bf) {
  int i = blockIdx.x * blockDim.x + threadIdx.x;   // B*H
  float v = h0[i];
  h[i] = v;
  h_bf[i] = f2bf(v);
}

// ---------------------------------------------------------------------------
// Attention: energy = tanh(enc_proj + dec_proj); logits = energy . v_att + b_v;
// masked softmax over S; context = alphas . enc. Also builds x = [y_t, context]
// in bf16 for the gi GEMM. One block (256 thr) per batch row.
// ---------------------------------------------------------------------------
__global__ void attention_kernel(const float* __restrict__ enc_proj,
                                 const float* __restrict__ dec_proj,
                                 const float* __restrict__ v_att,
                                 const float* __restrict__ b_v,
                                 const unsigned char* __restrict__ enc_mask,
                                 const float* __restrict__ enc,
                                 const unsigned short* __restrict__ emb_x, int t,
                                 float* __restrict__ context,
                                 unsigned short* __restrict__ x_bf)
{
  const int b = blockIdx.x, tid = threadIdx.x;       // 256 threads
  __shared__ float part[256];
  __shared__ float logits[S];
  __shared__ float alphas[S];

  const int s = tid & 63, q = tid >> 6;              // 4 threads per s
  const float* ep = enc_proj + ((long)b * S + s) * A;
  const float* dp = dec_proj + (long)b * A;
  float acc = 0.f;
  for (int a = q * 256; a < q * 256 + 256; ++a)
    acc += v_att[a] * tanhf(ep[a] + dp[a]);
  part[tid] = acc;
  __syncthreads();
  if (tid < S) {
    float l = part[s] + part[64 + s] + part[128 + s] + part[192 + s] + b_v[0];
    logits[s] = enc_mask[b * S + s] ? l : -INFINITY;
  }
  __syncthreads();
  if (tid == 0) {
    float m = -INFINITY;
    for (int i = 0; i < S; ++i) m = fmaxf(m, logits[i]);
    float sum = 0.f;
    for (int i = 0; i < S; ++i) { float e = __expf(logits[i] - m); alphas[i] = e; sum += e; }
    float inv = 1.f / sum;
    for (int i = 0; i < S; ++i) alphas[i] *= inv;
  }
  __syncthreads();
  for (int hh = tid; hh < H; hh += 256) {
    float c = 0.f;
    const float* eb = enc + (long)b * S * H + hh;
    for (int ss = 0; ss < S; ++ss) c += alphas[ss] * eb[ss * H];
    context[b * H + hh] = c;
    x_bf[(long)b * 2 * H + hh]     = emb_x[((long)b * T + t) * H + hh];  // y_t
    x_bf[(long)b * 2 * H + H + hh] = f2bf(c);                            // context
  }
}

// ---------------------------------------------------------------------------
// GRU gate fusion + state update + [h, context] bf16 concat for out_w GEMM.
// ---------------------------------------------------------------------------
__global__ void gru_combine(const float* __restrict__ gi, const float* __restrict__ gh,
                            const float* __restrict__ context,
                            const int* __restrict__ fra_length, int t,
                            float* __restrict__ h,
                            unsigned short* __restrict__ h_bf,
                            unsigned short* __restrict__ hc_bf)
{
  const int i = blockIdx.x * blockDim.x + threadIdx.x;   // B*H
  const int b = i >> 10, j = i & (H - 1);
  const float* gib = gi + (long)b * 3 * H;
  const float* ghb = gh + (long)b * 3 * H;
  float r = 1.f / (1.f + __expf(-(gib[j] + ghb[j])));
  float z = 1.f / (1.f + __expf(-(gib[H + j] + ghb[H + j])));
  float n = tanhf(gib[2 * H + j] + r * ghb[2 * H + j]);
  float hv = h[i];
  float hn = (1.f - z) * n + z * hv;
  float out = (t < fra_length[b]) ? hn : hv;
  h[i]    = out;
  h_bf[i] = f2bf(out);
  hc_bf[(long)b * 2 * H + j]     = f2bf(out);
  hc_bf[(long)b * 2 * H + H + j] = f2bf(context[i]);
}

// ---------------------------------------------------------------------------
// log_softmax over V per batch row, write to d_out[b, t, :].
// ---------------------------------------------------------------------------
__global__ void log_softmax_kernel(const float* __restrict__ logits,
                                   float* __restrict__ out, int t)
{
  const int b = blockIdx.x, tid = threadIdx.x;      // 256 threads
  __shared__ float red[256];
  const float* x = logits + (long)b * V;
  float m = -INFINITY;
  for (int v = tid; v < V; v += 256) m = fmaxf(m, x[v]);
  red[tid] = m; __syncthreads();
  for (int o = 128; o > 0; o >>= 1) { if (tid < o) red[tid] = fmaxf(red[tid], red[tid + o]); __syncthreads(); }
  m = red[0]; __syncthreads();
  float s = 0.f;
  for (int v = tid; v < V; v += 256) s += __expf(x[v] - m);
  red[tid] = s; __syncthreads();
  for (int o = 128; o > 0; o >>= 1) { if (tid < o) red[tid] += red[tid + o]; __syncthreads(); }
  const float lse = m + logf(red[0]);
  float* op = out + ((long)b * T + t) * V;
  for (int v = tid; v < V; v += 256) op[v] = x[v] - lse;
}

// ---------------------------------------------------------------------------
extern "C" void kernel_launch(void* const* d_in, const int* in_sizes, int n_in,
                              void* d_out, int out_size, void* d_ws, size_t ws_size,
                              hipStream_t stream) {
  const float* enc      = (const float*)d_in[0];
  const float* h0       = (const float*)d_in[1];
  const unsigned char* mask = (const unsigned char*)d_in[2];
  const int*   target   = (const int*)d_in[3];
  const int*   fra_len  = (const int*)d_in[4];
  const float* emb      = (const float*)d_in[5];
  const float* W_ih     = (const float*)d_in[6];
  const float* W_hh     = (const float*)d_in[7];
  const float* b_ih     = (const float*)d_in[8];
  const float* b_hh     = (const float*)d_in[9];
  const float* w_att    = (const float*)d_in[10];
  const float* b_att    = (const float*)d_in[11];
  const float* v_att    = (const float*)d_in[12];
  const float* b_v      = (const float*)d_in[13];
  const float* out_w_W  = (const float*)d_in[14];
  const float* out_w_b  = (const float*)d_in[15];
  const float* out_v_W  = (const float*)d_in[16];
  const float* out_v_b  = (const float*)d_in[17];
  float* out = (float*)d_out;

  // ---- workspace layout (all offsets 256B aligned) ----
  char* ws = (char*)d_ws; size_t off = 0;
  auto alloc = [&](size_t bytes) -> char* {
    char* p = ws + off; off = (off + bytes + 255) & ~(size_t)255; return p;
  };
  typedef unsigned short u16;
  u16* enc_bf  = (u16*)alloc((size_t)B * S * H * 2);
  u16* emb_x   = (u16*)alloc((size_t)B * T * H * 2);
  u16* We_bf   = (u16*)alloc((size_t)A * H * 2);
  u16* Wd_bf   = (u16*)alloc((size_t)A * H * 2);
  u16* Wih_bf  = (u16*)alloc((size_t)3 * H * 2 * H * 2);
  u16* Whh_bf  = (u16*)alloc((size_t)3 * H * H * 2);
  u16* oww_bf  = (u16*)alloc((size_t)H * 2 * H * 2);
  u16* ovw_bf  = (u16*)alloc((size_t)V * H * 2);
  u16* h_bf    = (u16*)alloc((size_t)B * H * 2);
  u16* x_bf    = (u16*)alloc((size_t)B * 2 * H * 2);
  u16* hc_bf   = (u16*)alloc((size_t)B * 2 * H * 2);
  u16* pre_bf  = (u16*)alloc((size_t)B * H * 2);
  float* enc_proj = (float*)alloc((size_t)B * S * A * 4);
  float* dec_proj = (float*)alloc((size_t)B * A * 4);
  float* gi       = (float*)alloc((size_t)B * 3 * H * 4);
  float* gh       = (float*)alloc((size_t)B * 3 * H * 4);
  float* h_f      = (float*)alloc((size_t)B * H * 4);
  float* ctx      = (float*)alloc((size_t)B * H * 4);
  float* pre_f    = (float*)alloc((size_t)B * H * 4);
  float* logits   = (float*)alloc((size_t)B * V * 4);
  (void)ws_size; (void)n_in; (void)in_sizes; (void)out_size;

  auto blocks = [](long n) { return (unsigned)((n + 255) / 256); };

  // ---- one-time prep ----
  cvt_bf16<<<blocks((long)B * S * H), 256, 0, stream>>>(enc, enc_bf, (long)B * S * H);
  gather_emb<<<blocks((long)B * T * H), 256, 0, stream>>>(emb, target, emb_x);
  split_watt<<<blocks((long)A * H), 256, 0, stream>>>(w_att, We_bf, Wd_bf);
  cvt_bf16<<<blocks((long)3 * H * 2 * H), 256, 0, stream>>>(W_ih, Wih_bf, (long)3 * H * 2 * H);
  cvt_bf16<<<blocks((long)3 * H * H), 256, 0, stream>>>(W_hh, Whh_bf, (long)3 * H * H);
  cvt_bf16<<<blocks((long)H * 2 * H), 256, 0, stream>>>(out_w_W, oww_bf, (long)H * 2 * H);
  cvt_bf16<<<blocks((long)V * H), 256, 0, stream>>>(out_v_W, ovw_bf, (long)V * H);
  init_h<<<blocks((long)B * H), 256, 0, stream>>>(h0, h_f, h_bf);

  // enc_proj[B*S, A] = enc_bf @ We^T + b_att   (M=2048, N=1024, K=1024)
  gemm_bf16_wmma<<<dim3(A / (16 * NT), (B * S) / (16 * MT)), 32, 0, stream>>>(
      enc_bf, H, We_bf, H, b_att, enc_proj, A, nullptr, H, 0);

  // ---- decode loop ----
  for (int t = 0; t < T; ++t) {
    // dec_proj = h @ Wd^T                        (M=32, N=1024, K=1024)
    gemm_bf16_wmma<<<dim3(A / (16 * NT), B / (16 * MT)), 32, 0, stream>>>(
        h_bf, H, Wd_bf, H, nullptr, dec_proj, A, nullptr, H, 0);
    // attention + softmax + context + x=[y_t,ctx]
    attention_kernel<<<B, 256, 0, stream>>>(enc_proj, dec_proj, v_att, b_v, mask,
                                            enc, emb_x, t, ctx, x_bf);
    // gi = x @ W_ih^T + b_ih                     (M=32, N=3072, K=2048)
    gemm_bf16_wmma<<<dim3((3 * H) / (16 * NT), B / (16 * MT)), 32, 0, stream>>>(
        x_bf, 2 * H, Wih_bf, 2 * H, b_ih, gi, 3 * H, nullptr, 2 * H, 0);
    // gh = h @ W_hh^T + b_hh                     (M=32, N=3072, K=1024)
    gemm_bf16_wmma<<<dim3((3 * H) / (16 * NT), B / (16 * MT)), 32, 0, stream>>>(
        h_bf, H, Whh_bf, H, b_hh, gh, 3 * H, nullptr, H, 0);
    // GRU gates + masked state update + [h,ctx] concat
    gru_combine<<<blocks((long)B * H), 256, 0, stream>>>(gi, gh, ctx, fra_len, t,
                                                         h_f, h_bf, hc_bf);
    // pre = tanh([h,ctx] @ out_w_W^T + b)        (M=32, N=1024, K=2048), bf16 copy
    gemm_bf16_wmma<<<dim3(H / (16 * NT), B / (16 * MT)), 32, 0, stream>>>(
        hc_bf, 2 * H, oww_bf, 2 * H, out_w_b, pre_f, H, pre_bf, 2 * H, 1);
    // logits = pre @ out_v_W^T + out_v_b         (M=32, N=32000, K=1024)
    gemm_bf16_wmma<<<dim3(V / (16 * NT), B / (16 * MT)), 32, 0, stream>>>(
        pre_bf, H, ovw_bf, H, out_v_b, logits, V, nullptr, H, 0);
    // log_softmax -> d_out[:, t, :]
    log_softmax_kernel<<<B, 256, 0, stream>>>(logits, out, t);
  }
}
